// Qwen3MoeMLP_20383914787230
// MI455X (gfx1250) — matrix operand order repaired and checked by reading, not compiled, not run on hardware
//
#include <hip/hip_runtime.h>

// ---------------------------------------------------------------------------
// Fused SwiGLU MLP for MI455X (gfx1250, wave32, WMMA bf16 -> f32 accum)
//   T=8192 tokens, H=2048 hidden, I=768 intermediate
//   out = down( silu(x @ gate^T) * (x @ up^T) )
//
// v2: register-blocked. Each wave owns the whole 32-row M tile (2 A frags)
//     and iterates N-tiles in pairs (phase A) / quads (phase B), pushing the
//     load:wmma ratio from 3:1 down to 1.5:1.
// ---------------------------------------------------------------------------

#define TT 8192
#define HH 2048
#define II 768
#define LDSS 776              // 768 + 8 bf16 pad -> rows start on distinct banks

typedef unsigned int u32;
typedef u32   u32x4   __attribute__((ext_vector_type(4)));
typedef __bf16 bf16x16 __attribute__((ext_vector_type(16)));
typedef __bf16 bf16x8  __attribute__((ext_vector_type(8)));
typedef float  f32x8   __attribute__((ext_vector_type(8)));

struct Frag128x2 { u32x4 lo, hi; };

// Build a 16-element bf16 fragment from two 16-byte chunks.
static __device__ __forceinline__ bf16x16 make_frag(const __bf16* plo,
                                                    const __bf16* phi) {
    Frag128x2 r;
    r.lo = *(const u32x4*)plo;
    r.hi = *(const u32x4*)phi;
    return __builtin_bit_cast(bf16x16, r);
}

static __device__ __forceinline__ f32x8 wmma_bf16(bf16x16 a, bf16x16 b, f32x8 c) {
    return __builtin_amdgcn_wmma_f32_16x16x32_bf16(
        /*neg_a=*/false, a, /*neg_b=*/false, b,
        /*c_mod=*/(short)0, c, /*reuse_a=*/false, /*reuse_b=*/false);
}

// ---------------------------------------------------------------------------
// Pass 0: f32 -> bf16 conversion (8 elems / thread)
// ---------------------------------------------------------------------------
__global__ void cvt_f32_to_bf16(const float* __restrict__ src,
                                __bf16* __restrict__ dst, int n) {
    int i = (blockIdx.x * blockDim.x + threadIdx.x) * 8;
    if (i >= n) return;
    float4 a = *(const float4*)(src + i);
    float4 b = *(const float4*)(src + i + 4);
    bf16x8 v;
    v[0] = (__bf16)a.x; v[1] = (__bf16)a.y; v[2] = (__bf16)a.z; v[3] = (__bf16)a.w;
    v[4] = (__bf16)b.x; v[5] = (__bf16)b.y; v[6] = (__bf16)b.z; v[7] = (__bf16)b.w;
    *(bf16x8*)(dst + i) = v;
}

// ---------------------------------------------------------------------------
// Pass 1: fused SwiGLU MLP. Grid = T/32 WGs, 256 threads = 8 waves.
//   Every wave covers the full 32 token rows (2 WMMA A-frags, M0/M1) and an
//   exclusive N strip: 96 cols of I in phase A, 256 cols of H in phase B.
// ---------------------------------------------------------------------------
__global__ void
__launch_bounds__(256)
swiglu_mlp_fused(const __bf16* __restrict__ xb,   // [T,H] row-major
                 const __bf16* __restrict__ gb,   // [I,H] row-major (== B^T)
                 const __bf16* __restrict__ ub,   // [I,H]
                 const __bf16* __restrict__ db,   // [H,I]
                 float* __restrict__ out)         // [T,H]
{
    __shared__ __align__(16) __bf16 hsh[32 * LDSS];

    const int lane = threadIdx.x & 31;
    const int wn   = threadIdx.x >> 5;   // wave id == N-strip id (8 strips)
    const int lh   = lane >> 4;          // lane half (selects K sub-chunk)
    const int l16  = lane & 15;

    const int m_g = blockIdx.x * 32;     // global token row base of this WG

    // A-fragment row pointers (M block 0: rows 0..15, M block 1: rows 16..31)
    const __bf16* arow0 = xb + (size_t)(m_g + l16) * HH;
    const __bf16* arow1 = xb + (size_t)(m_g + 16 + l16) * HH;

    // ---------------- Phase A: h = silu(x@gate^T) * (x@up^T) into LDS ------
    // per wave: 32 rows x 96 cols of I, in 3 outer steps of 2 N-tiles.
    for (int nt = 0; nt < 3; ++nt) {
        const int n0 = wn * 96 + nt * 32;          // column in I (2 tiles)
        f32x8 g00 = {}, g01 = {}, g10 = {}, g11 = {};
        f32x8 u00 = {}, u01 = {}, u10 = {}, u11 = {};
        const __bf16* grow0 = gb + (size_t)(n0 + l16) * HH;
        const __bf16* grow1 = gb + (size_t)(n0 + 16 + l16) * HH;
        const __bf16* urow0 = ub + (size_t)(n0 + l16) * HH;
        const __bf16* urow1 = ub + (size_t)(n0 + 16 + l16) * HH;
        for (int k0 = 0; k0 < HH; k0 += 32) {
            const __bf16* ap0 = arow0 + k0 + lh * 16;  // A: 16 contiguous K per lane
            const __bf16* ap1 = arow1 + k0 + lh * 16;
            bf16x16 a0 = make_frag(ap0, ap0 + 8);
            bf16x16 a1 = make_frag(ap1, ap1 + 8);
            const __bf16* gp0 = grow0 + k0 + lh * 16;  // B: 16 contiguous K per lane
            const __bf16* gp1 = grow1 + k0 + lh * 16;
            const __bf16* up0 = urow0 + k0 + lh * 16;
            const __bf16* up1 = urow1 + k0 + lh * 16;
            bf16x16 bg0 = make_frag(gp0, gp0 + 8);
            bf16x16 bg1 = make_frag(gp1, gp1 + 8);
            bf16x16 bu0 = make_frag(up0, up0 + 8);
            bf16x16 bu1 = make_frag(up1, up1 + 8);
            g00 = wmma_bf16(a0, bg0, g00);
            g10 = wmma_bf16(a1, bg0, g10);
            g01 = wmma_bf16(a0, bg1, g01);
            g11 = wmma_bf16(a1, bg1, g11);
            u00 = wmma_bf16(a0, bu0, u00);
            u10 = wmma_bf16(a1, bu0, u10);
            u01 = wmma_bf16(a0, bu1, u01);
            u11 = wmma_bf16(a1, bu1, u11);
        }
        // silu(gate)*up -> bf16 -> LDS h tile
        #pragma unroll
        for (int v = 0; v < 8; ++v) {
            const int r0 = v + lh * 8;          // row in M block 0
            const int r1 = 16 + v + lh * 8;     // row in M block 1
            float s;
            s = g00[v]; hsh[r0 * LDSS + n0 + l16]      = (__bf16)((s / (1.f + __expf(-s))) * u00[v]);
            s = g01[v]; hsh[r0 * LDSS + n0 + 16 + l16] = (__bf16)((s / (1.f + __expf(-s))) * u01[v]);
            s = g10[v]; hsh[r1 * LDSS + n0 + l16]      = (__bf16)((s / (1.f + __expf(-s))) * u10[v]);
            s = g11[v]; hsh[r1 * LDSS + n0 + 16 + l16] = (__bf16)((s / (1.f + __expf(-s))) * u11[v]);
        }
    }

    __syncthreads();

    // ---------------- Phase B: out = h @ down^T ----------------------------
    // per wave: 32 rows x 256 cols of H, in 4 outer steps of 4 N-tiles.
    const __bf16* lrow0 = &hsh[l16 * LDSS];          // LDS A-frag rows, M block 0
    const __bf16* lrow1 = &hsh[(16 + l16) * LDSS];   // M block 1
    for (int nt = 0; nt < 4; ++nt) {
        const int n0 = wn * 256 + nt * 64;     // column in H (4 tiles)
        f32x8 c00 = {}, c01 = {}, c02 = {}, c03 = {};
        f32x8 c10 = {}, c11 = {}, c12 = {}, c13 = {};
        const __bf16* brow0 = db + (size_t)(n0 + l16) * II;
        const __bf16* brow1 = db + (size_t)(n0 + 16 + l16) * II;
        const __bf16* brow2 = db + (size_t)(n0 + 32 + l16) * II;
        const __bf16* brow3 = db + (size_t)(n0 + 48 + l16) * II;
        for (int k0 = 0; k0 < II; k0 += 32) {
            const __bf16* ap0 = lrow0 + k0 + lh * 16;  // A fragments from LDS
            const __bf16* ap1 = lrow1 + k0 + lh * 16;
            bf16x16 a0 = make_frag(ap0, ap0 + 8);
            bf16x16 a1 = make_frag(ap1, ap1 + 8);
            const __bf16* bp0 = brow0 + k0 + lh * 16;
            const __bf16* bp1 = brow1 + k0 + lh * 16;
            const __bf16* bp2 = brow2 + k0 + lh * 16;
            const __bf16* bp3 = brow3 + k0 + lh * 16;
            bf16x16 b0 = make_frag(bp0, bp0 + 8);
            bf16x16 b1 = make_frag(bp1, bp1 + 8);
            bf16x16 b2 = make_frag(bp2, bp2 + 8);
            bf16x16 b3 = make_frag(bp3, bp3 + 8);
            c00 = wmma_bf16(a0, b0, c00);
            c10 = wmma_bf16(a1, b0, c10);
            c01 = wmma_bf16(a0, b1, c01);
            c11 = wmma_bf16(a1, b1, c11);
            c02 = wmma_bf16(a0, b2, c02);
            c12 = wmma_bf16(a1, b2, c12);
            c03 = wmma_bf16(a0, b3, c03);
            c13 = wmma_bf16(a1, b3, c13);
        }
        #pragma unroll
        for (int v = 0; v < 8; ++v) {
            size_t t0 = (size_t)(m_g + v + lh * 8);        // M block 0 global row
            size_t t1 = (size_t)(m_g + 16 + v + lh * 8);   // M block 1 global row
            out[t0 * HH + n0 + l16]      = c00[v];
            out[t0 * HH + n0 + 16 + l16] = c01[v];
            out[t0 * HH + n0 + 32 + l16] = c02[v];
            out[t0 * HH + n0 + 48 + l16] = c03[v];
            out[t1 * HH + n0 + l16]      = c10[v];
            out[t1 * HH + n0 + 16 + l16] = c11[v];
            out[t1 * HH + n0 + 32 + l16] = c12[v];
            out[t1 * HH + n0 + 48 + l16] = c13[v];
        }
    }
}

// ---------------------------------------------------------------------------
extern "C" void kernel_launch(void* const* d_in, const int* in_sizes, int n_in,
                              void* d_out, int out_size, void* d_ws, size_t ws_size,
                              hipStream_t stream) {
    const float* x      = (const float*)d_in[0];   // [T,H]
    const float* gate_w = (const float*)d_in[1];   // [I,H]
    const float* up_w   = (const float*)d_in[2];   // [I,H]
    const float* down_w = (const float*)d_in[3];   // [H,I]
    float* out = (float*)d_out;

    const int XN = TT * HH;   // 16,777,216
    const int WN = II * HH;   //  1,572,864

    __bf16* xb = (__bf16*)d_ws;
    __bf16* gb = xb + XN;
    __bf16* ub = gb + WN;
    __bf16* db = ub + WN;

    // pass 0: convert everything to bf16 (all counts divisible by 8*256)
    cvt_f32_to_bf16<<<XN / 8 / 256, 256, 0, stream>>>(x, xb, XN);
    cvt_f32_to_bf16<<<WN / 8 / 256, 256, 0, stream>>>(gate_w, gb, WN);
    cvt_f32_to_bf16<<<WN / 8 / 256, 256, 0, stream>>>(up_w, ub, WN);
    cvt_f32_to_bf16<<<WN / 8 / 256, 256, 0, stream>>>(down_w, db, WN);

    // pass 1: fused SwiGLU MLP, 32 tokens per WG
    swiglu_mlp_fused<<<TT / 32, 256, 0, stream>>>(xb, gb, ub, db, out);
}
